// Attention_128849019324
// MI455X (gfx1250) — matrix operand order repaired and checked by reading, not compiled, not run on hardware
//
#include <hip/hip_runtime.h>

// ---------------------------------------------------------------------------
// MI455X (gfx1250, wave32) fused attention, round 2:
//  prep_kernel: W (f32, [K][N]) -> Wt (f16, [N][K]) once, so GEMM B-frags are
//               two contiguous 16B loads instead of 16 scalar load+cvt.
//  proj_kernel: one wave computes a 16x64 output tile (whole head) -> A-frag
//               loaded/converted once per K-step, 4 WMMAs per step.
//  attn_kernel: flash attention, 64-key blocks, K tile staged via the CDNA5
//               Tensor Data Mover (tensor_load_to_lds + s_wait_tensorcnt),
//               V staged transposed for contiguous PV B-fragments.
// ---------------------------------------------------------------------------

typedef __attribute__((ext_vector_type(16))) _Float16 v16h;
typedef __attribute__((ext_vector_type(8)))  _Float16 v8h;
typedef __attribute__((ext_vector_type(8)))  float    v8f;
typedef __attribute__((ext_vector_type(4)))  unsigned int u32x4;
typedef __attribute__((ext_vector_type(8)))  int      i32x8;
typedef __attribute__((ext_vector_type(4)))  int      i32x4;

#define NB 4
#define NS 2048
#define NDM 1024
#define NH 16
#define ND 64

#if defined(__has_builtin)
#if __has_builtin(__builtin_amdgcn_tensor_load_to_lds) && \
    __has_builtin(__builtin_amdgcn_s_wait_tensorcnt)
#define HAVE_TDM 1
#endif
#endif

static __device__ __forceinline__ v16h cat8(v8h a, v8h b) {
  return __builtin_shufflevector(a, b, 0,1,2,3,4,5,6,7,8,9,10,11,12,13,14,15);
}
static __device__ __forceinline__ v8h ld8(const _Float16* p) {
  return *(const v8h*)p;
}
static __device__ __forceinline__ float redmax16(float x) {
  #pragma unroll
  for (int m = 1; m < 16; m <<= 1) x = fmaxf(x, __shfl_xor(x, m, 32));
  return x;
}
static __device__ __forceinline__ float redsum16(float x) {
  #pragma unroll
  for (int m = 1; m < 16; m <<= 1) x += __shfl_xor(x, m, 32);
  return x;
}

// ---------------------------------------------------------------------------
// Kernel 0: Wt[n*1024+k] = (f16)W[k*1024+n].  grid=(4096,3), block=256.
// ---------------------------------------------------------------------------
__global__ __launch_bounds__(256) void prep_kernel(
    const float* __restrict__ Wq, const float* __restrict__ Wk,
    const float* __restrict__ Wv, _Float16* __restrict__ Wt)
{
  const float* W = (blockIdx.y == 0) ? Wq : (blockIdx.y == 1) ? Wk : Wv;
  _Float16* O = Wt + (size_t)blockIdx.y * NDM * NDM;
  int idx = blockIdx.x * 256 + threadIdx.x;   // n*1024 + k, write-coalesced
  int n = idx >> 10, k = idx & 1023;
  O[idx] = (_Float16)W[(size_t)k * NDM + n];
}

// ---------------------------------------------------------------------------
// Kernel 1: Y[b][h][s][d] (f16) = X @ W. One wave per 16x64 tile (one head).
// grid = (1024, 3), block = 256 (8 waves).
// ---------------------------------------------------------------------------
__global__ __launch_bounds__(256) void proj_kernel(
    const float* __restrict__ Xq, const float* __restrict__ Xk, const float* __restrict__ Xv,
    const _Float16* __restrict__ Wt,
    _Float16* __restrict__ Yq, _Float16* __restrict__ Yk, _Float16* __restrict__ Yv)
{
  const float* X; const _Float16* Wm; _Float16* Y;
  if (blockIdx.y == 0)      { X = Xq; Wm = Wt;                 Y = Yq; }
  else if (blockIdx.y == 1) { X = Xk; Wm = Wt + NDM * NDM;     Y = Yk; }
  else                      { X = Xv; Wm = Wt + 2 * NDM * NDM; Y = Yv; }

  const int wave = threadIdx.x >> 5;
  const int lane = threadIdx.x & 31;
  const int tile  = blockIdx.x * 8 + wave;   // 8192 tiles = 512 Mtiles x 16 heads
  const int mtile = tile >> 4;
  const int head  = tile & 15;

  const int l15 = lane & 15;
  const int hiA = (lane & 16) ? 8 : 0;       // A-frag K swizzle base
  const int hiB = (lane & 16) ? 8 : 0;       // B-frag K base
  const int row = mtile * 16 + l15;

  v8f acc0 = {}, acc1 = {}, acc2 = {}, acc3 = {};
  for (int kb = 0; kb < NDM; kb += 32) {
    // A fragment: 16x32 slab of X, per-lane K = {kb+hiA+0..7, kb+hiA+16..23}
    const float* ap = X + (size_t)row * NDM + kb + hiA;
    float4 f0 = *(const float4*)(ap);
    float4 f1 = *(const float4*)(ap + 4);
    float4 f2 = *(const float4*)(ap + 16);
    float4 f3 = *(const float4*)(ap + 20);
    v16h a;
    a[0]=(_Float16)f0.x; a[1]=(_Float16)f0.y; a[2]=(_Float16)f0.z; a[3]=(_Float16)f0.w;
    a[4]=(_Float16)f1.x; a[5]=(_Float16)f1.y; a[6]=(_Float16)f1.z; a[7]=(_Float16)f1.w;
    a[8]=(_Float16)f2.x; a[9]=(_Float16)f2.y; a[10]=(_Float16)f2.z; a[11]=(_Float16)f2.w;
    a[12]=(_Float16)f3.x; a[13]=(_Float16)f3.y; a[14]=(_Float16)f3.z; a[15]=(_Float16)f3.w;

    // 4 B fragments from transposed f16 weights: two 8-half spans each, K = kb+hiB+{0..7, 16..23}
    #pragma unroll
    for (int nc = 0; nc < 4; ++nc) {
      const _Float16* wp = Wm + (size_t)(head * 64 + nc * 16 + l15) * NDM + kb + hiB;
      v16h bw = cat8(ld8(wp), ld8(wp + 16));
      if (nc == 0) acc0 = __builtin_amdgcn_wmma_f32_16x16x32_f16(false, a, false, bw, (short)0, acc0, false, false);
      if (nc == 1) acc1 = __builtin_amdgcn_wmma_f32_16x16x32_f16(false, a, false, bw, (short)0, acc1, false, false);
      if (nc == 2) acc2 = __builtin_amdgcn_wmma_f32_16x16x32_f16(false, a, false, bw, (short)0, acc2, false, false);
      if (nc == 3) acc3 = __builtin_amdgcn_wmma_f32_16x16x32_f16(false, a, false, bw, (short)0, acc3, false, false);
    }
  }

  // Store D tiles (C-layout: lanes 0-15 rows 0-7, lanes 16-31 rows 8-15)
  const int b = mtile >> 7, sbase = (mtile & 127) * 16;
  const int rhalf = (lane & 16) ? 8 : 0;
  #pragma unroll
  for (int v = 0; v < 8; ++v) {
    int s = sbase + v + rhalf;
    _Float16* yp = Y + (((size_t)b * NH + head) * NS + s) * ND + l15;
    yp[0]  = (_Float16)acc0[v];
    yp[16] = (_Float16)acc1[v];
    yp[32] = (_Float16)acc2[v];
    yp[48] = (_Float16)acc3[v];
  }
}

// ---------------------------------------------------------------------------
// Kernel 2: flash attention, 64-key blocks. One block per (b, h, 128 q-rows);
// 8 waves each own a 16-row query tile. grid = (16, 16, 4), block = 256.
// ---------------------------------------------------------------------------
__global__ __launch_bounds__(256) void attn_kernel(
    const _Float16* __restrict__ Qw, const _Float16* __restrict__ Kw,
    const _Float16* __restrict__ Vw,
    const int* __restrict__ Qlen, const int* __restrict__ Vlen,
    float* __restrict__ Out)
{
  __shared__ _Float16 Kt[64 * 64];        // [kpos][hd] row-major (TDM target)
  __shared__ _Float16 Vt[64 * 64];        // transposed [hd][kpos]
  __shared__ _Float16 Pt[8][16 * 64];     // per-wave P tile (C-layout -> A-layout)

  const int wave = threadIdx.x >> 5;
  const int lane = threadIdx.x & 31;
  const int b = blockIdx.z, h = blockIdx.y;
  const int qbase = blockIdx.x * 128 + wave * 16;

  const int vlen = Vlen[b];
  const int qlen = Qlen[b];

  const int l15 = lane & 15;
  const int hiA = (lane & 16) ? 8 : 0;
  const int hiB = (lane & 16) ? 8 : 0;
  const int rhalf = (lane & 16) ? 8 : 0;

  const _Float16* Qp = Qw + (((size_t)b * NH + h) * NS + qbase) * ND;
  const _Float16* Kp = Kw + (((size_t)b * NH + h) * NS) * ND;
  const _Float16* Vp = Vw + (((size_t)b * NH + h) * NS) * ND;

  // Q A-fragments, held in registers for the whole key loop
  const v16h qa0 = cat8(ld8(Qp + l15 * ND + hiA),      ld8(Qp + l15 * ND + hiA + 16));
  const v16h qa1 = cat8(ld8(Qp + l15 * ND + 32 + hiA), ld8(Qp + l15 * ND + 32 + hiA + 16));

  v8f o0 = {}, o1 = {}, o2 = {}, o3 = {};
  float mi[8], li[8];
  #pragma unroll
  for (int v = 0; v < 8; ++v) { mi[v] = -__builtin_inff(); li[v] = 0.f; }

  const float scale = 0.125f;  // 1/sqrt(64)

  for (int kb = 0; kb < NS / 64; ++kb) {
    __syncthreads();   // previous iteration's LDS reads complete

#ifdef HAVE_TDM
    if (wave == 0) {
      // K tile = 64 rows x 64 halves = 8KB contiguous -> 1-D TDM descriptor,
      // data_size=3 (8B elements), tile_dim0 = tensor_dim0 = 1024.
      unsigned ldsoff = (unsigned)(uintptr_t)&Kt[0];
      unsigned long long ga =
          (unsigned long long)(uintptr_t)(Kp + (size_t)kb * 64 * ND);
      u32x4 g0 = { 1u,                                   // count=1, user mode
                   ldsoff,                               // lds_addr (bytes)
                   (unsigned)(ga & 0xffffffffu),         // global_addr[31:0]
                   (unsigned)((ga >> 32) & 0x1ffffffu) | (2u << 30) }; // type=2
      i32x8 g1 = { (int)(3u << 16),        // data_size=3 (8B), mask=0
                   (int)(1024u << 16),     // tensor_dim0[15:0] = 1024
                   (int)(1u << 16),        // tensor_dim1 = 1
                   (int)(1024u << 16),     // tile_dim0 = 1024
                   0,                      // tile_dim1 = tile_dim2 = 0
                   1024,                   // tensor_dim0_stride = 1024
                   0, 0 };
      i32x4 z4 = { 0, 0, 0, 0 };
#if defined(__clang_major__) && (__clang_major__ >= 23)
      i32x8 z8 = { 0, 0, 0, 0, 0, 0, 0, 0 };
      __builtin_amdgcn_tensor_load_to_lds(g0, g1, z4, z4, z8, 0);
#else
      __builtin_amdgcn_tensor_load_to_lds(g0, g1, z4, z4, 0);
#endif
    }
#else
    {   // fallback: cooperative K staging (row-major)
      #pragma unroll
      for (int it = 0; it < 2; ++it) {
        int lin = (threadIdx.x + it * 256) * 8;
        int kp = lin >> 6, hd = lin & 63;
        *(v8h*)&Kt[kp * 64 + hd] = ld8(Kp + (size_t)(kb * 64 + kp) * ND + hd);
      }
    }
#endif
    {   // cooperative V staging, transposed: Vt[hd][kpos]
      #pragma unroll
      for (int it = 0; it < 2; ++it) {
        int lin = (threadIdx.x + it * 256) * 8;
        int kp = lin >> 6, hd = lin & 63;
        v8h vv = ld8(Vp + (size_t)(kb * 64 + kp) * ND + hd);
        #pragma unroll
        for (int e = 0; e < 8; ++e) Vt[(hd + e) * 64 + kp] = vv[e];
      }
    }
#ifdef HAVE_TDM
    if (wave == 0) __builtin_amdgcn_s_wait_tensorcnt(0);
#endif
    __syncthreads();   // Kt (TDM) + Vt staged and visible

    // ---- S = Q @ K^T : four 16x16 score frags (keys f*16 .. f*16+15) ----
    v8f sf[4];
    #pragma unroll
    for (int f = 0; f < 4; ++f) {
      int kr = (f * 16 + l15) * 64;
      v16h bk0 = cat8(ld8(&Kt[kr + hiB]),      ld8(&Kt[kr + hiB + 16]));
      v16h bk1 = cat8(ld8(&Kt[kr + 32 + hiB]), ld8(&Kt[kr + 32 + hiB + 16]));
      v8f s = {};
      s = __builtin_amdgcn_wmma_f32_16x16x32_f16(false, qa0, false, bk0, (short)0, s, false, false);
      s = __builtin_amdgcn_wmma_f32_16x16x32_f16(false, qa1, false, bk1, (short)0, s, false, false);
      sf[f] = s;
    }

    // ---- online softmax in C-layout; write P tile to LDS as it is made ----
    const int key0 = kb * 64 + l15;
    const float sub0 = (key0      >= vlen) ? 1.0e12f : 0.f;
    const float sub1 = (key0 + 16 >= vlen) ? 1.0e12f : 0.f;
    const float sub2 = (key0 + 32 >= vlen) ? 1.0e12f : 0.f;
    const float sub3 = (key0 + 48 >= vlen) ? 1.0e12f : 0.f;
    _Float16* Pw = Pt[wave];
    float alpha[8];
    #pragma unroll
    for (int v = 0; v < 8; ++v) {
      float x0 = sf[0][v] * scale - sub0;
      float x1 = sf[1][v] * scale - sub1;
      float x2 = sf[2][v] * scale - sub2;
      float x3 = sf[3][v] * scale - sub3;
      float mrow = redmax16(fmaxf(fmaxf(x0, x1), fmaxf(x2, x3)));
      float mnew = fmaxf(mi[v], mrow);
      float a  = __expf(mi[v] - mnew);
      float e0 = __expf(x0 - mnew);
      float e1 = __expf(x1 - mnew);
      float e2 = __expf(x2 - mnew);
      float e3 = __expf(x3 - mnew);
      li[v] = li[v] * a + redsum16((e0 + e1) + (e2 + e3));
      mi[v] = mnew;
      alpha[v] = a;
      int m = v + rhalf;
      Pw[m * 64 + l15]      = (_Float16)e0;
      Pw[m * 64 + 16 + l15] = (_Float16)e1;
      Pw[m * 64 + 32 + l15] = (_Float16)e2;
      Pw[m * 64 + 48 + l15] = (_Float16)e3;
    }
    #pragma unroll
    for (int v = 0; v < 8; ++v) {
      o0[v] *= alpha[v]; o1[v] *= alpha[v]; o2[v] *= alpha[v]; o3[v] *= alpha[v];
    }

    // ---- O += P @ V : P as two A-frags (kpos 0..31 / 32..63), V^T B-frags
    v16h pa0 = cat8(ld8(&Pw[l15 * 64 + hiA]),      ld8(&Pw[l15 * 64 + hiA + 16]));
    v16h pa1 = cat8(ld8(&Pw[l15 * 64 + 32 + hiA]), ld8(&Pw[l15 * 64 + 32 + hiA + 16]));
    #pragma unroll
    for (int nc = 0; nc < 4; ++nc) {
      int vr = (nc * 16 + l15) * 64;
      v16h bv0 = cat8(ld8(&Vt[vr + hiB]),      ld8(&Vt[vr + hiB + 16]));
      v16h bv1 = cat8(ld8(&Vt[vr + 32 + hiB]), ld8(&Vt[vr + 32 + hiB + 16]));
      v8f* op = (nc == 0) ? &o0 : (nc == 1) ? &o1 : (nc == 2) ? &o2 : &o3;
      *op = __builtin_amdgcn_wmma_f32_16x16x32_f16(false, pa0, false, bv0, (short)0, *op, false, false);
      *op = __builtin_amdgcn_wmma_f32_16x16x32_f16(false, pa1, false, bv1, (short)0, *op, false, false);
    }
  }

  // ---- epilogue: normalize, query-length mask, store f32 ----
  #pragma unroll
  for (int v = 0; v < 8; ++v) {
    int m = v + rhalf;
    int s = qbase + m;
    float inv = (s < qlen) ? (1.0f / li[v]) : 0.0f;
    size_t base = ((size_t)b * NS + s) * NDM + (size_t)h * ND;
    Out[base +  0 + l15] = o0[v] * inv;
    Out[base + 16 + l15] = o1[v] * inv;
    Out[base + 32 + l15] = o2[v] * inv;
    Out[base + 48 + l15] = o3[v] * inv;
  }
}

// ---------------------------------------------------------------------------
extern "C" void kernel_launch(void* const* d_in, const int* in_sizes, int n_in,
                              void* d_out, int out_size, void* d_ws, size_t ws_size,
                              hipStream_t stream) {
  (void)in_sizes; (void)n_in; (void)out_size; (void)ws_size;
  const float* Qs = (const float*)d_in[0];
  const float* Ks = (const float*)d_in[1];
  const float* Vs = (const float*)d_in[2];
  const int*   Ql = (const int*)  d_in[3];
  const int*   Vl = (const int*)  d_in[4];
  const float* WQ = (const float*)d_in[5];
  const float* WK = (const float*)d_in[6];
  const float* WV = (const float*)d_in[7];
  float* out = (float*)d_out;

  const size_t elems = (size_t)NB * NH * NS * ND;   // 8,388,608 halves each
  _Float16* qws = (_Float16*)d_ws;                  // 16 MB
  _Float16* kws = qws + elems;                      // 16 MB
  _Float16* vws = kws + elems;                      // 16 MB
  _Float16* wts = vws + elems;                      // 6 MB (3x 1024x1024 f16)

  prep_kernel<<<dim3(4096, 3), 256, 0, stream>>>(WQ, WK, WV, wts);
  proj_kernel<<<dim3(1024, 3), 256, 0, stream>>>(Qs, Ks, Vs, wts, qws, kws, vws);
  attn_kernel<<<dim3(NS / 128, NH, NB), 256, 0, stream>>>(qws, kws, vws,
                                                          Ql, Vl, out);
}
